// Layers_RNN_88536455840384
// MI455X (gfx1250) — compile-verified, matrix operand order checked by reading, not run on hardware
//
#include <hip/hip_runtime.h>

typedef __attribute__((ext_vector_type(4)))  float    v4f;
typedef __attribute__((ext_vector_type(8)))  float    v8f;
typedef __attribute__((ext_vector_type(4)))  __bf16   v4bf;
typedef __attribute__((ext_vector_type(8)))  __bf16   v8bf;
typedef __attribute__((ext_vector_type(16))) __bf16   v16bf;
typedef __attribute__((ext_vector_type(4)))  unsigned v4u;
typedef __attribute__((ext_vector_type(8)))  unsigned v8u;

#define LSEQ 512
#define BB   64
#define HH   1024
#define MTOT (LSEQ * BB)     // 32768 rows in the projection GEMMs
#define NWG_SCAN 16          // persistent workgroups in the recurrent scan
#define NS  (HH / NWG_SCAN)  // 64 W_hh rows (output cols) per WG -> 128KB LDS slice

// Build a 16x32 bf16 WMMA fragment from K-contiguous memory.
// ISA A-layout: lanes 0-15 hold K[ks..ks+8) and K[16+ks..24+ks), ks=(lane>>4)*8.
// Caller pre-offsets p by (row*lda + ks).
__device__ __forceinline__ v16bf frag16(const __bf16* p) {
  v8bf lo = *(const v8bf*)(p);
  v8bf hi = *(const v8bf*)(p + 16);
  return __builtin_shufflevector(lo, hi, 0,1,2,3,4,5,6,7,8,9,10,11,12,13,14,15);
}

__device__ __forceinline__ v8f wmma_bf16(v16bf a, v16bf b, v8f c) {
  return __builtin_amdgcn_wmma_f32_16x16x32_bf16(false, a, false, b, (short)0, c,
                                                 false, false);
}

__device__ __forceinline__ int rfl(unsigned x) {
  return __builtin_amdgcn_readfirstlane((int)x);
}

__global__ void zero_u32(unsigned* p, int n) {
  int i = threadIdx.x;
  if (i < n) p[i] = 0u;
}

__global__ void f32_to_bf16(const float* __restrict__ in, __bf16* __restrict__ out,
                            int n4) {
  int i = blockIdx.x * blockDim.x + threadIdx.x;
  if (i < n4) {
    v4f v = *(const v4f*)(in + (size_t)i * 4);
    v4bf o;
    o[0] = (__bf16)v[0]; o[1] = (__bf16)v[1];
    o[2] = (__bf16)v[2]; o[3] = (__bf16)v[3];
    *(v4bf*)(out + (size_t)i * 4) = o;
  }
}

// out[m, n] = sum_k A[m,k] * W[n,k] + bias[n]
// A: MTOT x HH bf16 (K contiguous). W: HH x HH bf16 (K contiguous == column-major B).
// Grid: (MTOT/32, HH/256), block 256 (8 waves). Wave = 16(M) x 64(N), register
// double-buffered over K so WMMAs overlap the next iteration's global loads.
__global__ __launch_bounds__(256) void proj_gemm(const __bf16* __restrict__ A,
                                                 const __bf16* __restrict__ W,
                                                 const float* __restrict__ bias,
                                                 float* __restrict__ out) {
  const int tid  = threadIdx.x;
  const int w    = tid >> 5, lane = tid & 31;
  const int mbase = blockIdx.x * 32 + (w & 1) * 16;
  const int nbase = blockIdx.y * 256 + (w >> 1) * 64;
  const int m  = lane & 15;
  const int ks = (lane >> 4) * 8;
  const int nn = lane & 15;

  const __bf16* arow = A + (size_t)(mbase + m) * HH + ks;
  const __bf16* br0  = W + (size_t)(nbase +  0 + nn) * HH + ks;
  const __bf16* br1  = W + (size_t)(nbase + 16 + nn) * HH + ks;
  const __bf16* br2  = W + (size_t)(nbase + 32 + nn) * HH + ks;
  const __bf16* br3  = W + (size_t)(nbase + 48 + nn) * HH + ks;

  v8f c0 = {}, c1 = {}, c2 = {}, c3 = {};
  v16bf a  = frag16(arow);
  v16bf b0 = frag16(br0), b1 = frag16(br1), b2 = frag16(br2), b3 = frag16(br3);
  for (int k = 0; k < HH - 32; k += 32) {
    v16bf an  = frag16(arow + k + 32);          // prefetch next K-slab into regs
    v16bf bn0 = frag16(br0 + k + 32);
    v16bf bn1 = frag16(br1 + k + 32);
    v16bf bn2 = frag16(br2 + k + 32);
    v16bf bn3 = frag16(br3 + k + 32);
    c0 = wmma_bf16(a, b0, c0);
    c1 = wmma_bf16(a, b1, c1);
    c2 = wmma_bf16(a, b2, c2);
    c3 = wmma_bf16(a, b3, c3);
    a = an; b0 = bn0; b1 = bn1; b2 = bn2; b3 = bn3;
  }
  c0 = wmma_bf16(a, b0, c0);
  c1 = wmma_bf16(a, b1, c1);
  c2 = wmma_bf16(a, b2, c2);
  c3 = wmma_bf16(a, b3, c3);

  const int mrow = mbase + ((lane >> 4) << 3);
  v8f cc[4] = {c0, c1, c2, c3};
#pragma unroll
  for (int j = 0; j < 4; ++j) {
    const int ncol = nbase + j * 16 + nn;
    const float bv = bias[ncol];
#pragma unroll
    for (int r = 0; r < 8; ++r)
      out[(size_t)(mrow + r) * HH + ncol] = cc[j][r] + bv;
  }
}

// Persistent recurrent scan: h_{t+1} = tanh(xp_t + h_t @ Whh^T).
// 16 WGs; each WG keeps W_hh rows [wg*64, wg*64+64) resident in LDS for all 512
// steps (staged once by the Tensor Data Mover), and async-copies the full h_t
// (128KB bf16) into LDS each step via global_load_async_to_lds_b128.
// Dynamic LDS: 128KB sW + 128KB sH = 256KB (< 320KB per WGP).
__global__ __launch_bounds__(256) void rnn_scan(const float* __restrict__ xp,
                                                const __bf16* __restrict__ Whh,
                                                __bf16* __restrict__ hbuf,
                                                const float* __restrict__ h0,
                                                __bf16* out_bf, float* out_f32,
                                                float* __restrict__ hn_out,
                                                unsigned* __restrict__ ctr) {
  extern __shared__ char smem[];
  __bf16* sW = (__bf16*)smem;                    // NS x HH = 128KB
  __bf16* sH = (__bf16*)(smem + NS * HH * 2);    // BB x HH = 128KB
  const int wg  = blockIdx.x;
  const int tid = threadIdx.x;

  // ---- Stage W_hh slice once via TDM (2D tile: dim0=1024 cols, dim1=64 rows) ----
  {
    const unsigned long long gaddr =
        (unsigned long long)(uintptr_t)(Whh + (size_t)wg * NS * HH);
    const unsigned lds_w = (unsigned)(uintptr_t)(void*)sW;
    v4u g0;
    g0[0] = rfl(1u);                                   // count=1, user mode
    g0[1] = rfl(lds_w);                                // lds_addr
    g0[2] = rfl((unsigned)gaddr);                      // global_addr[31:0]
    g0[3] = rfl((unsigned)(gaddr >> 32) | (2u << 30)); // addr[56:32] | type=2
    v8u g1;
    g1[0] = rfl(1u << 16);        // data_size=1 (2 bytes)
    g1[1] = rfl(1024u << 16);     // tensor_dim0 = 1024 (bits 79:48)
    g1[2] = rfl(1024u << 16);     // tensor_dim1 = 1024 (bits 111:80)
    g1[3] = rfl(1024u << 16);     // tile_dim0 = 1024 (bits 127:112)
    g1[4] = rfl(64u);             // tile_dim1 = 64
    g1[5] = rfl(1024u);           // tensor_dim0_stride = 1024
    g1[6] = rfl(0u);
    g1[7] = rfl(0u);
    if (tid < 32) {               // one wave issues the DMA (TDM ignores EXEC)
      asm volatile("tensor_load_to_lds %0, %1" :: "s"(g0), "s"(g1) : "memory");
      __builtin_amdgcn_s_wait_tensorcnt(0);
    }
  }
  // Convert this WG's slice of h0 into bf16 ping buffer 0 (global).
  for (int i = tid; i < (BB * HH) / NWG_SCAN; i += 256) {
    int g = wg * ((BB * HH) / NWG_SCAN) + i;
    hbuf[g] = (__bf16)h0[g];
  }
  __syncthreads();

  auto gbar = [&](unsigned expected) {
    __builtin_amdgcn_fence(__ATOMIC_RELEASE, "agent");
    __syncthreads();
    if (tid == 0) {
      __hip_atomic_fetch_add(ctr, 1u, __ATOMIC_RELAXED, __HIP_MEMORY_SCOPE_AGENT);
      while (__hip_atomic_load(ctr, __ATOMIC_RELAXED, __HIP_MEMORY_SCOPE_AGENT) <
             expected)
        __builtin_amdgcn_s_sleep(1);
    }
    __syncthreads();
    __builtin_amdgcn_fence(__ATOMIC_ACQUIRE, "agent");
  };

  gbar(NWG_SCAN);  // h0 conversion visible device-wide

  const int w     = tid >> 5;
  const int lane  = tid & 31;
  const int mbase = (w & 3) * 16;          // 4 M-tiles cover batch 64
  const int nloc0 = (w >> 2) * 32;         // wave owns 2 N-tiles: nloc0, nloc0+16
  const int m  = lane & 15;
  const int ks = (lane >> 4) * 8;
  const int nn = lane & 15;
  const unsigned sH_off = (unsigned)(uintptr_t)(void*)sH;
  const __bf16* bbase0 = sW + (size_t)(nloc0 +  0 + nn) * HH + ks;
  const __bf16* bbase1 = sW + (size_t)(nloc0 + 16 + nn) * HH + ks;
  const __bf16* abase  = sH + (size_t)(mbase + m) * HH + ks;

  for (int t = 0; t < LSEQ; ++t) {
    const __bf16* hc = hbuf + (size_t)(t & 1) * (BB * HH);
    __bf16*       hn = hbuf + (size_t)((t + 1) & 1) * (BB * HH);
    const float* xpt = xp + (size_t)t * (BB * HH);
    __builtin_prefetch(xpt + (size_t)mbase * HH + wg * NS + nloc0 + nn, 0, 0);

    // Async-copy h_t (64x1024 bf16, 128KB) global -> LDS: 8192 b128 chunks.
    for (int i = tid; i < (BB * HH) / 8; i += 256) {
      unsigned loff = sH_off + (unsigned)i * 16u;
      unsigned long long gsrc = (unsigned long long)(uintptr_t)(hc + (size_t)i * 8);
      asm volatile("global_load_async_to_lds_b128 %0, %1, off"
                   :: "v"(loff), "v"(gsrc) : "memory");
    }
    asm volatile("s_wait_asynccnt 0x0" ::: "memory");
    __syncthreads();

    v8f c0 = {}, c1 = {};
    for (int k = 0; k < HH; k += 32) {
      v16bf av = frag16(abase + k);
      c0 = wmma_bf16(av, frag16(bbase0 + k), c0);
      c1 = wmma_bf16(av, frag16(bbase1 + k), c1);
    }

    const int mrow = mbase + ((lane >> 4) << 3);
    const int ng0 = wg * NS + nloc0 + nn;
    v8f cc[2] = {c0, c1};
#pragma unroll
    for (int j = 0; j < 2; ++j) {
      const int ng = ng0 + j * 16;
#pragma unroll
      for (int r = 0; r < 8; ++r) {
        const size_t o = (size_t)(mrow + r) * HH + ng;
        const float v = tanhf(cc[j][r] + xpt[o]);
        hn[o] = (__bf16)v;
        if (out_bf)  out_bf[(size_t)t * (BB * HH) + o] = (__bf16)v;
        if (out_f32) out_f32[(size_t)t * (BB * HH) + o] = v;
        if (t == LSEQ - 1) hn_out[o] = v;
      }
    }
    __syncthreads();  // all ds reads of sH done before next step's async overwrite
    gbar((unsigned)NWG_SCAN * (unsigned)(t + 2));
  }
}

extern "C" void kernel_launch(void* const* d_in, const int* in_sizes, int n_in,
                              void* d_out, int out_size, void* d_ws, size_t ws_size,
                              hipStream_t stream) {
  (void)in_sizes; (void)n_in; (void)out_size; (void)ws_size;
  const float* x    = (const float*)d_in[0];
  const float* h0   = (const float*)d_in[1];
  const float* Wih0 = (const float*)d_in[2];
  const float* Whh0 = (const float*)d_in[3];
  const float* b0   = (const float*)d_in[4];
  const float* Wih1 = (const float*)d_in[5];
  const float* Whh1 = (const float*)d_in[6];
  const float* b1   = (const float*)d_in[7];
  float* out = (float*)d_out;

  char* ws = (char*)d_ws;
  __bf16*   wbf  = (__bf16*)(ws);                        // 4 x 1M bf16 weights (8MB)
  __bf16*   hbuf = (__bf16*)(ws + ((size_t)8 << 20));    // 2 x 64K bf16 h ping-pong
  unsigned* ctr  = (unsigned*)(ws + ((size_t)9 << 20));  // barrier counters
  float*    xp   = (float*)(ws + ((size_t)16 << 20));    // 32768x1024 f32 (128MB)
  __bf16*   abf  = (__bf16*)(ws + ((size_t)144 << 20));  // 32768x1024 bf16 (64MB)

  __bf16* wih0b = wbf + ((size_t)0 << 20);
  __bf16* whh0b = wbf + ((size_t)1 << 20);
  __bf16* wih1b = wbf + ((size_t)2 << 20);
  __bf16* whh1b = wbf + ((size_t)3 << 20);

  // Idempotent host-side attribute set; not a stream op (capture-safe), no guards.
  hipFuncSetAttribute((const void*)rnn_scan,
                      hipFuncAttributeMaxDynamicSharedMemorySize, 262144);

  zero_u32<<<1, 64, 0, stream>>>(ctr, 64);

  const int nw4 = (HH * HH) / 4;
  f32_to_bf16<<<nw4 / 256, 256, 0, stream>>>(Wih0, wih0b, nw4);
  f32_to_bf16<<<nw4 / 256, 256, 0, stream>>>(Whh0, whh0b, nw4);
  f32_to_bf16<<<nw4 / 256, 256, 0, stream>>>(Wih1, wih1b, nw4);
  f32_to_bf16<<<nw4 / 256, 256, 0, stream>>>(Whh1, whh1b, nw4);
  const int nx4 = (MTOT * HH) / 4;
  f32_to_bf16<<<nx4 / 256, 256, 0, stream>>>(x, abf, nx4);

  dim3 pg(MTOT / 32, HH / 256);
  // Layer 0: projection then scan (scan writes bf16 out0 over abf; x dead by then)
  proj_gemm<<<pg, 256, 0, stream>>>(abf, wih0b, b0, xp);
  rnn_scan<<<NWG_SCAN, 256, 262144, stream>>>(xp, whh0b, hbuf, h0, abf, nullptr,
                                              out + (size_t)MTOT * HH, ctr + 0);
  // Layer 1: projection of out0 then scan writing fp32 out1 + h_n[1]
  proj_gemm<<<pg, 256, 0, stream>>>(abf, wih1b, b1, xp);
  rnn_scan<<<NWG_SCAN, 256, 262144, stream>>>(xp, whh1b, hbuf, h0 + BB * HH,
                                              nullptr, out,
                                              out + (size_t)MTOT * HH + BB * HH,
                                              ctr + 1);
}
